// ManualLSTM_69930657513969
// MI455X (gfx1250) — matrix operand order repaired and checked by reading, not compiled, not run on hardware
//
#include <hip/hip_runtime.h>
#include <hip/hip_bf16.h>
#include <math.h>

typedef __bf16 bf16;
typedef __attribute__((ext_vector_type(8)))  __bf16 v8bf;
typedef __attribute__((ext_vector_type(16))) __bf16 v16bf;
typedef __attribute__((ext_vector_type(8)))  float  v8f;

#define LSTM_B 32
#define LSTM_S 2048
#define LSTM_I 256
#define LSTM_H 512
#define LSTM_G 2048   // 4*H

static __device__ __forceinline__ float sigmoidf_fast(float z) {
    return 1.0f / (1.0f + __expf(-z));
}

// ---------------------------------------------------------------------------
// Prep: convert weights to bf16 (then resident in 192MB L2 for the scan)
// ---------------------------------------------------------------------------
__global__ void lstm_prep_weights(const float* __restrict__ Wih,
                                  const float* __restrict__ Whh,
                                  bf16* __restrict__ Wih_b,
                                  bf16* __restrict__ Whh_b) {
    const int n1 = LSTM_G * LSTM_I;
    const int n2 = LSTM_G * LSTM_H;
    for (int idx = blockIdx.x * blockDim.x + threadIdx.x; idx < n1 + n2;
         idx += gridDim.x * blockDim.x) {
        if (idx < n1) Wih_b[idx]      = (bf16)Wih[idx];
        else          Whh_b[idx - n1] = (bf16)Whh[idx - n1];
    }
}

__global__ void lstm_prep_state(const float* __restrict__ h0,
                                const float* __restrict__ c0,
                                bf16* __restrict__ hb0,
                                float* __restrict__ c_ws) {
    int i = blockIdx.x * blockDim.x + threadIdx.x;
    if (i < LSTM_B * LSTM_H) {
        hb0[i]  = (bf16)h0[i];
        c_ws[i] = c0[i];
    }
}

// A-fragment loader for fp32 source rows (convert on the fly).
// Contiguous k order used here (B matches): lanes 0-15 hold K {0..15}, lanes 16-31 hold {16..31}.
static __device__ __forceinline__ v16bf load_a_f32(const float* ap) {
    v8f lo = *(const v8f*)(ap);
    v8f up = *(const v8f*)(ap + 8);
    v16bf a;
    #pragma unroll
    for (int e = 0; e < 8; ++e) { a[e] = (bf16)lo[e]; a[e + 8] = (bf16)up[e]; }
    return a;
}

// A fragments (2 M-tiles) from fp32 x rows at timestep t, k-block k0.
static __device__ __forceinline__ void load_ax(const float* __restrict__ x,
                                               int t, int k0, int kA, int col,
                                               v16bf a[2]) {
    #pragma unroll
    for (int m = 0; m < 2; ++m) {
        const int b = m * 16 + col;
        a[m] = load_a_f32(x + ((size_t)b * LSTM_S + t) * LSTM_I + k0 + kA);
    }
}

// A fragments (2 M-tiles) from bf16 h rows at k-block k0.
static __device__ __forceinline__ void load_ah(const bf16* __restrict__ h_in,
                                               int k0, int kA, int col,
                                               v16bf a[2]) {
    #pragma unroll
    for (int m = 0; m < 2; ++m) {
        const int b = m * 16 + col;
        const bf16* ap = h_in + (size_t)b * LSTM_H + k0 + kA;
        v8bf lo = *(const v8bf*)(ap);
        v8bf up = *(const v8bf*)(ap + 8);
        a[m] = __builtin_shufflevector(lo, up,
            0, 1, 2, 3, 4, 5, 6, 7, 8, 9, 10, 11, 12, 13, 14, 15);
    }
}

// B fragments for the 4 gates at gate-columns (q*H + n), k-block k0.
static __device__ __forceinline__ void load_b4(const bf16* __restrict__ W,
                                               size_t ld, int n, int k0, int kB,
                                               v16bf b[4]) {
    #pragma unroll
    for (int q = 0; q < 4; ++q)
        b[q] = *(const v16bf*)(W + (size_t)(q * LSTM_H + n) * ld + k0 + kB);
}

// ---------------------------------------------------------------------------
// Full-chip input-projection GEMM: xp[t, g, b] = sum_i x[b,t,i] * Wih[g,i].
// Output layout (S, 4H, B): one v8f per lane == exact WMMA C-fragment layout
// (VGPR r <-> M=r / r+8 <-> consecutive b), so the scan kernel seeds its
// accumulators with plain v8f loads. Grid (32 g-blocks, S); 128 thr = 4 waves.
// ---------------------------------------------------------------------------
__global__ __launch_bounds__(128, 2) void lstm_xproj(
    const float* __restrict__ x,      // (B,S,I) fp32
    const bf16*  __restrict__ Wih_b,  // (4H,I)  bf16
    float*       __restrict__ xp)     // (S,4H,B) fp32
{
    const int lane = threadIdx.x & 31;
    const int wib  = threadIdx.x >> 5;              // wave in block: 0..3
    const int col  = lane & 15;
    const int hi   = lane >> 4;
    const int g0   = blockIdx.x * 64 + wib * 16;    // gate-column tile start
    const int t    = blockIdx.y;

    const int kA = hi * 16;
    const int kB = hi * 16;

    v8f acc[2];
    #pragma unroll
    for (int m = 0; m < 2; ++m)
        #pragma unroll
        for (int e = 0; e < 8; ++e) acc[m][e] = 0.0f;

    // depth-1 pipelined, fully unrolled (8 k-steps)
    v16bf aCur[2], bCur;
    load_ax(x, t, 0, kA, col, aCur);
    bCur = *(const v16bf*)(Wih_b + (size_t)(g0 + col) * LSTM_I + kB);
    #pragma unroll
    for (int k0 = 0; k0 < LSTM_I; k0 += 32) {
        v16bf aNxt[2], bNxt;
        if (k0 + 32 < LSTM_I) {
            load_ax(x, t, k0 + 32, kA, col, aNxt);
            bNxt = *(const v16bf*)(Wih_b + (size_t)(g0 + col) * LSTM_I + k0 + 32 + kB);
        }
        #pragma unroll
        for (int m = 0; m < 2; ++m)
            acc[m] = __builtin_amdgcn_wmma_f32_16x16x32_bf16(
                false, aCur[m], false, bCur, (short)0, acc[m], false, false);
        if (k0 + 32 < LSTM_I) {
            aCur[0] = aNxt[0]; aCur[1] = aNxt[1]; bCur = bNxt;
        }
    }
    #pragma unroll
    for (int m = 0; m < 2; ++m)
        *(v8f*)(xp + ((size_t)t * LSTM_G + (g0 + col)) * LSTM_B + m * 16 + hi * 8) = acc[m];
}

// ---------------------------------------------------------------------------
// One LSTM timestep. XP=true: accumulators seeded from precomputed x_proj,
// K=512 (Whh only). XP=false: fused, K=768 ([x_t,h] @ [Wih,Whh]^T).
// Grid: 8 blocks x 128 threads = 32 waves; wave w owns gate columns
// [16w,16w+16) of all 4 gates (same H-slice -> local gate combine).
// k-loops are fully unrolled with depth-1 software pipelining: next k-step's
// 6 fragment loads are in flight while the current 8 WMMAs execute.
// ---------------------------------------------------------------------------
template <bool XP>
__global__ __launch_bounds__(128, 1) void lstm_step(
    const float* __restrict__ x,      // (B,S,I) fp32
    const float* __restrict__ xp,     // (S,4H,B) fp32 (XP path)
    const bf16*  __restrict__ Wih_b,  // (4H,I)  bf16
    const bf16*  __restrict__ Whh_b,  // (4H,H)  bf16
    const float* __restrict__ bias,   // (4H,)   fp32
    const bf16*  __restrict__ h_in,   // (B,H)   bf16
    bf16*        __restrict__ h_out,  // (B,H)   bf16
    float*       __restrict__ c_st,   // (B,H)   fp32 in/out
    float*       __restrict__ out,    // (B,S,H) fp32
    int t)
{
    const int lane   = threadIdx.x & 31;
    const int waveid = (int)((blockIdx.x * blockDim.x + threadIdx.x) >> 5);
    const int n0     = waveid * 16;      // H-slice start
    const int col    = lane & 15;
    const int hi     = lane >> 4;
    const int n      = n0 + col;

    const int kA = hi * 16;
    const int kB = hi * 16;

    v8f acc[4][2];
    if constexpr (XP) {
        // Seed accumulators with the precomputed input projection (layout match).
        #pragma unroll
        for (int q = 0; q < 4; ++q)
            #pragma unroll
            for (int m = 0; m < 2; ++m)
                acc[q][m] = *(const v8f*)(xp +
                    ((size_t)t * LSTM_G + (q * LSTM_H + n)) * LSTM_B + m * 16 + hi * 8);
    } else {
        #pragma unroll
        for (int q = 0; q < 4; ++q)
            #pragma unroll
            for (int m = 0; m < 2; ++m)
                #pragma unroll
                for (int e = 0; e < 8; ++e) acc[q][m][e] = 0.0f;

        // Input projection fused onto the serial path (8 k-steps, pipelined).
        v16bf aCur[2], bCur[4];
        load_ax(x, t, 0, kA, col, aCur);
        load_b4(Wih_b, LSTM_I, n, 0, kB, bCur);
        #pragma unroll
        for (int k0 = 0; k0 < LSTM_I; k0 += 32) {
            v16bf aNxt[2], bNxt[4];
            if (k0 + 32 < LSTM_I) {
                load_ax(x, t, k0 + 32, kA, col, aNxt);
                load_b4(Wih_b, LSTM_I, n, k0 + 32, kB, bNxt);
            }
            #pragma unroll
            for (int q = 0; q < 4; ++q)
                #pragma unroll
                for (int m = 0; m < 2; ++m)
                    acc[q][m] = __builtin_amdgcn_wmma_f32_16x16x32_bf16(
                        false, aCur[m], false, bCur[q], (short)0, acc[q][m], false, false);
            if (k0 + 32 < LSTM_I) {
                aCur[0] = aNxt[0]; aCur[1] = aNxt[1];
                #pragma unroll
                for (int q = 0; q < 4; ++q) bCur[q] = bNxt[q];
            }
        }
    }

    // Recurrent projection, K = H (16 k-steps, pipelined, fully unrolled).
    {
        v16bf aCur[2], bCur[4];
        load_ah(h_in, 0, kA, col, aCur);
        load_b4(Whh_b, LSTM_H, n, 0, kB, bCur);
        #pragma unroll
        for (int k0 = 0; k0 < LSTM_H; k0 += 32) {
            v16bf aNxt[2], bNxt[4];
            if (k0 + 32 < LSTM_H) {
                load_ah(h_in, k0 + 32, kA, col, aNxt);
                load_b4(Whh_b, LSTM_H, n, k0 + 32, kB, bNxt);
            }
            #pragma unroll
            for (int q = 0; q < 4; ++q)
                #pragma unroll
                for (int m = 0; m < 2; ++m)
                    acc[q][m] = __builtin_amdgcn_wmma_f32_16x16x32_bf16(
                        false, aCur[m], false, bCur[q], (short)0, acc[q][m], false, false);
            if (k0 + 32 < LSTM_H) {
                aCur[0] = aNxt[0]; aCur[1] = aNxt[1];
                #pragma unroll
                for (int q = 0; q < 4; ++q) bCur[q] = bNxt[q];
            }
        }
    }

    // Gate nonlinearity + state update.
    const float b_i = bias[0 * LSTM_H + n];
    const float b_f = bias[1 * LSTM_H + n];
    const float b_g = bias[2 * LSTM_H + n];
    const float b_o = bias[3 * LSTM_H + n];

    #pragma unroll
    for (int m = 0; m < 2; ++m) {
        #pragma unroll
        for (int r = 0; r < 8; ++r) {
            const int brow = m * 16 + hi * 8 + r;   // batch index
            const float iv = sigmoidf_fast(acc[0][m][r] + b_i);
            const float fv = sigmoidf_fast(acc[1][m][r] + b_f);
            const float gv = tanhf(acc[2][m][r] + b_g);
            const float ov = sigmoidf_fast(acc[3][m][r] + b_o);
            const size_t cidx = (size_t)brow * LSTM_H + n;
            const float cn = fv * c_st[cidx] + iv * gv;
            c_st[cidx] = cn;
            const float hv = ov * tanhf(cn);
            out[((size_t)brow * LSTM_S + t) * LSTM_H + n] = hv;
            h_out[cidx] = (bf16)hv;
        }
    }
}

// ---------------------------------------------------------------------------
// Write final h (== output[:, S-1, :]) and c into the tuple tail of d_out.
// ---------------------------------------------------------------------------
__global__ void lstm_finalize(const float* __restrict__ c_st,
                              float* __restrict__ out) {
    int i = blockIdx.x * blockDim.x + threadIdx.x;
    if (i < LSTM_B * LSTM_H) {
        const int b = i / LSTM_H, nn = i % LSTM_H;
        const size_t base = (size_t)LSTM_B * LSTM_S * LSTM_H;
        out[base + i] = out[((size_t)b * LSTM_S + (LSTM_S - 1)) * LSTM_H + nn];
        out[base + (size_t)LSTM_B * LSTM_H + i] = c_st[i];
    }
}

extern "C" void kernel_launch(void* const* d_in, const int* in_sizes, int n_in,
                              void* d_out, int out_size, void* d_ws, size_t ws_size,
                              hipStream_t stream) {
    (void)in_sizes; (void)n_in; (void)out_size;
    const float* x    = (const float*)d_in[0];
    const float* h0   = (const float*)d_in[1];
    const float* c0   = (const float*)d_in[2];
    const float* Wih  = (const float*)d_in[3];
    const float* bias = (const float*)d_in[4];
    const float* Whh  = (const float*)d_in[5];
    float* out = (float*)d_out;

    const size_t xp_bytes    = (size_t)LSTM_S * LSTM_G * LSTM_B * sizeof(float); // 512 MB
    const size_t small_bytes = (size_t)LSTM_G * LSTM_I * sizeof(bf16)
                             + (size_t)LSTM_G * LSTM_H * sizeof(bf16)
                             + (size_t)LSTM_B * LSTM_H * sizeof(bf16) * 2
                             + (size_t)LSTM_B * LSTM_H * sizeof(float);          // ~3.4 MB
    const bool fast = ws_size >= xp_bytes + small_bytes;

    char* ws = (char*)d_ws;
    float* xp = nullptr;
    if (fast) { xp = (float*)ws; ws += xp_bytes; }
    bf16* Wih_b = (bf16*)ws;  ws += (size_t)LSTM_G * LSTM_I * sizeof(bf16);
    bf16* Whh_b = (bf16*)ws;  ws += (size_t)LSTM_G * LSTM_H * sizeof(bf16);
    bf16* hb0   = (bf16*)ws;  ws += (size_t)LSTM_B * LSTM_H * sizeof(bf16);
    bf16* hb1   = (bf16*)ws;  ws += (size_t)LSTM_B * LSTM_H * sizeof(bf16);
    float* c_st = (float*)ws;

    lstm_prep_weights<<<1024, 256, 0, stream>>>(Wih, Whh, Wih_b, Whh_b);
    lstm_prep_state<<<(LSTM_B * LSTM_H + 255) / 256, 256, 0, stream>>>(h0, c0, hb0, c_st);

    bf16* hbuf[2] = { hb0, hb1 };
    if (fast) {
        lstm_xproj<<<dim3(32, LSTM_S), 128, 0, stream>>>(x, Wih_b, xp);
        for (int t = 0; t < LSTM_S; ++t)
            lstm_step<true><<<8, 128, 0, stream>>>(x, xp, Wih_b, Whh_b, bias,
                                                   hbuf[t & 1], hbuf[(t + 1) & 1],
                                                   c_st, out, t);
    } else {
        for (int t = 0; t < LSTM_S; ++t)
            lstm_step<false><<<8, 128, 0, stream>>>(x, nullptr, Wih_b, Whh_b, bias,
                                                    hbuf[t & 1], hbuf[(t + 1) & 1],
                                                    c_st, out, t);
    }
    lstm_finalize<<<(LSTM_B * LSTM_H + 255) / 256, 256, 0, stream>>>(c_st, out);
}
